// ExpertSelector_24713241821317
// MI455X (gfx1250) — matrix operand order repaired and checked by reading, not compiled, not run on hardware
//
#include <hip/hip_runtime.h>
#include <hip/hip_bf16.h>

// ---------------- problem constants (from reference setup) ----------------
#define NTOK 32768            // B*S = 8*4096
#define H_  1024
#define HH_ 512
#define E_  64
#define NCOLS 576             // 64 router cols + 512 confidence-hidden cols
#define MTILE 64              // tokens per block
#define KTILE 64              // K per LDS stage (2 WMMA K-steps of 32)
#define TPW 9                 // col tiles per wave (36 tiles / 4 col-groups)
#define WSTRIDE 72            // LDS row stride in bf16 (64 data + 8 pad; 144B, 16B-aligned)
#define NCHUNK (H_ / KTILE)   // 16
#define NTHREADS 512          // 16 waves

typedef __attribute__((ext_vector_type(16))) __bf16 v16bf;
typedef __attribute__((ext_vector_type(8)))  float  v8f;

// scheduling-group hints (args must be integer constant expressions)
// 0x008 = MFMA/WMMA group, 0x100 = DS-read group
#define SGB(mask, size, id) __builtin_amdgcn_sched_group_barrier((mask), (size), (id))

// float -> bf16 round-to-nearest-even in integer ops
__device__ __forceinline__ unsigned short f32_to_bf16(float f) {
    unsigned u = __float_as_uint(f);
    unsigned r = 0x7FFFu + ((u >> 16) & 1u);
    return (unsigned short)((u + r) >> 16);
}

// CDNA5 async copy, GVS form: SGPR 64-bit base + per-lane 32-bit byte offset.
__device__ __forceinline__ void async_copy_b128_s(unsigned lds_byte_addr,
                                                  unsigned goff_bytes,
                                                  unsigned long long sbase) {
    asm volatile("global_load_async_to_lds_b128 %0, %1, %2"
                 :: "v"(lds_byte_addr), "v"(goff_bytes), "s"(sbase)
                 : "memory");
}
__device__ __forceinline__ void wait_async0() {
    asm volatile("s_wait_asynccnt 0x0" ::: "memory");
}

// ---------------- kernel 0: zero confidence accumulator ----------------
__global__ void zero_conf(float* __restrict__ acc) {
    int i = blockIdx.x * blockDim.x + threadIdx.x;
    if (i < NTOK) acc[i] = 0.0f;
}

// ---------------- kernel 1: pack W_router + W1 into combined bf16 [576,1024] ----
__global__ void cvt_weights(const float* __restrict__ Wr,
                            const float* __restrict__ W1,
                            unsigned short* __restrict__ out) {
    int i = blockIdx.x * blockDim.x + threadIdx.x;
    if (i >= NCOLS * H_) return;
    int row = i >> 10;
    int k   = i & (H_ - 1);
    float v = (row < E_) ? Wr[row * H_ + k] : W1[(row - E_) * H_ + k];
    out[i] = f32_to_bf16(v);
}

// ---------------- kernel 2: hidden f32 -> bf16 (x4 vectorized) ----------------
__global__ void cvt_hidden(const float* __restrict__ in, unsigned short* __restrict__ out,
                           long n4) {
    long i = (long)blockIdx.x * blockDim.x + threadIdx.x;
    if (i >= n4) return;
    float4 f = ((const float4*)in)[i];
    ushort4 o;
    o.x = f32_to_bf16(f.x);
    o.y = f32_to_bf16(f.y);
    o.z = f32_to_bf16(f.z);
    o.w = f32_to_bf16(f.w);
    ((ushort4*)out)[i] = o;
}

// ---------------- kernel 3: fused GEMM, async-LDS double buffered ----------
__global__ __launch_bounds__(NTHREADS) void moe_gemm(
    const unsigned short* __restrict__ A,      // [NTOK, H] bf16 bits
    const unsigned short* __restrict__ W,      // [576, H] bf16 bits
    const float* __restrict__ b_router,
    const float* __restrict__ b1,
    const float* __restrict__ W2,
    float* __restrict__ out_logits,            // [NTOK, 64]
    float* __restrict__ conf_acc)              // [NTOK]
{
    __shared__ unsigned short sm[2][(NCOLS + MTILE) * WSTRIDE];   // 2 x 90KB

    const int tid  = threadIdx.x;
    const int wave = tid >> 5;
    const int lane = tid & 31;
    const int half = lane >> 4;
    const int l16  = lane & 15;
    const int r    = wave & 3;           // row subtile 0..3 (16 tokens each)
    const int cg   = wave >> 2;          // column group 0..3 (9 tiles each)
    const int tokBase = blockIdx.x * MTILE;
    const int rowTok  = tokBase + r * 16;

    // ---- fixed per-thread staging offsets (16B granules; KTILE slice = 128B/row) ----
    unsigned wGOff[9], wLOff[9];
#pragma unroll
    for (int j = 0; j < 9; ++j) {
        const int g  = j * NTHREADS + tid;
        const int wn = g >> 3;                       // weight row 0..575
        const int wp = g & 7;                        // 16B granule in 128B slice
        wGOff[j] = (unsigned)((wn * H_ + wp * 8) * 2);
        wLOff[j] = (unsigned)((wn * WSTRIDE + wp * 8) * 2);
    }
    const int an = tid >> 3, ap = tid & 7;
    const unsigned aGOff = (unsigned)((an * H_ + ap * 8) * 2);
    const unsigned aLOff = (unsigned)(((NCOLS + an) * WSTRIDE + ap * 8) * 2);

    // uniform 64-bit bases kept in SGPRs, advanced by 128B per chunk
    unsigned long long wBase = (unsigned long long)(size_t)W;
    unsigned long long aBase = (unsigned long long)(size_t)(A + (size_t)tokBase * H_);

    v8f zero = {};
    v8f acc[TPW];
#pragma unroll
    for (int j = 0; j < TPW; ++j) acc[j] = zero;

    auto stage = [&](int buf) {
        const unsigned smb = (unsigned)(size_t)(&sm[buf][0]);
#pragma unroll
        for (int j = 0; j < 9; ++j)
            async_copy_b128_s(smb + wLOff[j], wGOff[j], wBase);
        async_copy_b128_s(smb + aLOff, aGOff, aBase);
        wBase += KTILE * 2;
        aBase += KTILE * 2;
    };

    // consume one KTILE=64 chunk (two WMMA K-steps) from buffer `cur`
    auto compute = [&](const unsigned short* cur) {
        union F { v16bf v; uint4 q[2]; };
        const unsigned short* arow = cur + (NCOLS + r * 16 + l16) * WSTRIDE;
        const unsigned short* bb   = cur + (cg * (TPW * 16) + l16) * WSTRIDE + half * 16;
#pragma unroll
        for (int s = 0; s < 2; ++s) {                 // K sub-step: k_local = s*32 ..
            F a;
            {
                const unsigned short* ar = arow + s * 32 + half * 16;
                a.q[0] = *(const uint4*)(ar);
                a.q[1] = *(const uint4*)(ar + 8);
            }
            F b[3];
#pragma unroll
            for (int p = 0; p < 2; ++p) {
                b[p].q[0] = *(const uint4*)(bb + p * 16 * WSTRIDE + s * 32);
                b[p].q[1] = *(const uint4*)(bb + p * 16 * WSTRIDE + s * 32 + 8);
            }
#pragma unroll
            for (int j = 0; j < TPW; ++j) {
                if (j + 2 < TPW) {
                    const int q = (j + 2) % 3;
                    b[q].q[0] = *(const uint4*)(bb + (j + 2) * 16 * WSTRIDE + s * 32);
                    b[q].q[1] = *(const uint4*)(bb + (j + 2) * 16 * WSTRIDE + s * 32 + 8);
                }
                acc[j] = __builtin_amdgcn_wmma_f32_16x16x32_bf16(
                    false, a.v, false, b[j % 3].v, (short)0, acc[j], false, false);
            }
            // scheduling hint: software-pipeline DS reads 2 tiles ahead of WMMA.
            // 20 DS reads (A2 + B18) and 9 WMMAs per sub-step:
            //   DS(6) [A + tiles 0,1] ; { WMMA(1) ; DS(2) } x7 ; WMMA(2)
            SGB(0x100, 6, 0);
#pragma unroll
            for (int i = 0; i < 7; ++i) {
                SGB(0x008, 1, 0);
                SGB(0x100, 2, 0);
            }
            SGB(0x008, 2, 0);
        }
    };

    // ---- pipeline: double-buffered, compile-time buffer selection ----
    stage(0);
    wait_async0();
    __syncthreads();
    for (int c = 0; c < NCHUNK; c += 2) {
        stage(1);                        // chunk c+1 (c max 14 -> always valid)
        compute(&sm[0][0]);
        wait_async0();
        __syncthreads();
        if (c + 2 < NCHUNK) stage(0);
        compute(&sm[1][0]);
        wait_async0();
        __syncthreads();
    }

    // ---- epilogue. C layout: VGPR i, lanes 0-15 -> (M=i,N=l16); 16-31 -> (M=i+8,N=l16)
#pragma unroll
    for (int j = 0; j < TPW; ++j) {
        const int n0 = (cg * TPW + j) * 16;
        const int n  = n0 + l16;
        if (n0 < E_) {
            const float bias = b_router[n];
#pragma unroll
            for (int i = 0; i < 8; ++i) {
                const int m = rowTok + half * 8 + i;
                out_logits[(size_t)m * E_ + n] = acc[j][i] + bias;   // TEMPERATURE = 1
            }
        } else {
            const float bias = b1[n - E_];
            const float w2   = W2[n - E_];
#pragma unroll
            for (int i = 0; i < 8; ++i) {
                float hv = acc[j][i] + bias;
                hv = hv > 0.0f ? hv : 0.0f;
                float cc = hv * w2;
                cc += __shfl_xor(cc, 1, 32);
                cc += __shfl_xor(cc, 2, 32);
                cc += __shfl_xor(cc, 4, 32);
                cc += __shfl_xor(cc, 8, 32);
                if (l16 == 0) {
                    const int m = rowTok + half * 8 + i;
                    atomicAdd(&conf_acc[m], cc);
                }
            }
        }
    }
}

// ---------------- kernel 4: per-token softmax + dynamic top-k ----------------
__global__ __launch_bounds__(256) void moe_select(
    const float* __restrict__ conf_acc,
    const float* __restrict__ logits,
    const float* __restrict__ b2,
    float* __restrict__ sel_w,
    float* __restrict__ sel_i,
    float* __restrict__ conf_out)
{
    const int t = blockIdx.x * blockDim.x + threadIdx.x;
    if (t >= NTOK) return;

    const float conf = 1.0f / (1.0f + __expf(-(conf_acc[t] + b2[0])));
    conf_out[t] = conf;

    float dkf = rintf(1.0f + 7.0f * (1.0f - conf));   // RNE like jnp.round
    dkf = fminf(fmaxf(dkf, 1.0f), 8.0f);
    const int dk = (int)dkf;

    float v[E_];
    float mx = -3.402823466e+38f;
#pragma unroll
    for (int j = 0; j < E_; ++j) {
        v[j] = logits[(size_t)t * E_ + j];
        mx = fmaxf(mx, v[j]);
    }
    float s = 0.0f;
#pragma unroll
    for (int j = 0; j < E_; ++j) {
        v[j] = __expf(v[j] - mx);
        s += v[j];
    }
    const float inv = 1.0f / s;
#pragma unroll
    for (int j = 0; j < E_; ++j) v[j] *= inv;

    unsigned long long used = 0ull;
#pragma unroll
    for (int slot = 0; slot < 8; ++slot) {
        float bv = -1.0f;
        int   bi = 0;
#pragma unroll
        for (int j = 0; j < E_; ++j) {
            const bool free = ((used >> j) & 1ull) == 0ull;
            if (free && v[j] > bv) { bv = v[j]; bi = j; }  // strict '>' => lowest idx on ties
        }
        used |= (1ull << bi);
        const bool active = slot < dk;
        sel_w[(size_t)t * 8 + slot] = active ? bv : 0.0f;
        sel_i[(size_t)t * 8 + slot] = active ? (float)bi : 0.0f;
    }
}

// ---------------- launcher ----------------
extern "C" void kernel_launch(void* const* d_in, const int* in_sizes, int n_in,
                              void* d_out, int out_size, void* d_ws, size_t ws_size,
                              hipStream_t stream) {
    const float* hidden = (const float*)d_in[0];   // [8,4096,1024]
    const float* Wr     = (const float*)d_in[1];   // [64,1024]
    const float* br     = (const float*)d_in[2];   // [64]
    const float* W1     = (const float*)d_in[3];   // [512,1024]
    const float* b1     = (const float*)d_in[4];   // [512]
    const float* W2     = (const float*)d_in[5];   // [1,512]
    const float* b2     = (const float*)d_in[6];   // [1]

    // d_out (float32 view): sel_w [N,8] | sel_i [N,8] | confidence [N] | router_logits [N,64]
    float* out    = (float*)d_out;
    float* sel_w  = out;
    float* sel_i  = out + (size_t)NTOK * 8;
    float* conf   = out + (size_t)NTOK * 16;
    float* logits = out + (size_t)NTOK * 16 + NTOK;

    // workspace: A_bf16 [N,H] | W_bf16 [576,H] | conf_acc [N]
    char* ws = (char*)d_ws;
    unsigned short* Abf = (unsigned short*)ws;
    unsigned short* Wbf = (unsigned short*)(ws + (size_t)NTOK * H_ * 2);
    float* conf_acc     = (float*)(ws + (size_t)NTOK * H_ * 2 + (size_t)NCOLS * H_ * 2);

    zero_conf<<<(NTOK + 255) / 256, 256, 0, stream>>>(conf_acc);
    cvt_weights<<<(NCOLS * H_ + 255) / 256, 256, 0, stream>>>(Wr, W1, Wbf);
    {
        const long n4 = (long)NTOK * H_ / 4;
        cvt_hidden<<<(unsigned)((n4 + 255) / 256), 256, 0, stream>>>(hidden, Abf, n4);
    }
    moe_gemm<<<NTOK / MTILE, NTHREADS, 0, stream>>>(Abf, Wbf, br, b1, W2, logits, conf_acc);
    moe_select<<<(NTOK + 255) / 256, 256, 0, stream>>>(conf_acc, logits, b2,
                                                       sel_w, sel_i, conf);
}
